// Block_45037027066271
// MI455X (gfx1250) — matrix operand order repaired and checked by reading, not compiled, not run on hardware
//
#include <hip/hip_runtime.h>
#include <cmath>

typedef __attribute__((ext_vector_type(16))) _Float16 v16h;
typedef __attribute__((ext_vector_type(8)))  float    v8f;
typedef __attribute__((ext_vector_type(4)))  int      v4i;

#define C_DIM 768
#define NH    12
#define HD    64
#define B_    8
#define T_    8
#define HW_   196
#define NTOK  1569        // 1 + T*H*W
#define M1    12544       // temporal tokens  (B*HW*T)
#define M2    12608       // spatial tokens   (B*T*197)
#define M3    12552       // B*NTOK
#define QKVN  2304
#define FFN   3072

#if defined(__gfx1250__) && __has_builtin(__builtin_amdgcn_global_load_async_to_lds_b128) && __has_builtin(__builtin_amdgcn_s_wait_asynccnt)
#define USE_ASYNC 1
#else
#define USE_ASYNC 0
#endif

typedef __attribute__((address_space(1))) v4i as1_v4i;
typedef __attribute__((address_space(3))) v4i as3_v4i;
#define GP128(p) ((as1_v4i*)(p))
#define LP128(p) ((as3_v4i*)(p))

// ---------------------------------------------------------------- utilities

__global__ void cvt_f32_f16(const float* __restrict__ in, _Float16* __restrict__ out, int n) {
    for (int i = blockIdx.x * blockDim.x + threadIdx.x; i < n; i += gridDim.x * blockDim.x)
        out[i] = (_Float16)in[i];
}

__device__ __forceinline__ float block_sum256(float v) {
    __shared__ float partial[8];
    for (int off = 16; off > 0; off >>= 1) v += __shfl_xor(v, off, 32);
    int lane = threadIdx.x & 31, w = threadIdx.x >> 5;
    __syncthreads();                 // protect reuse across calls
    if (lane == 0) partial[w] = v;
    __syncthreads();
    float r = partial[0];
    #pragma unroll
    for (int i = 1; i < 8; ++i) r += partial[i];
    return r;
}

// -------------------------------------------------- gather + LayerNorm rows
// mode 0: temporal gather from x   (row = bhw*T + t)   -> also stash raw fp32
// mode 1: spatial gather: j==0 -> cls from x, else from xt3 (row = bt*197+j)
// mode 2: plain contiguous rows of x
__global__ __launch_bounds__(256)
void ln_rows_kernel(const float* __restrict__ x, const float* __restrict__ alt,
                    const float* __restrict__ g, const float* __restrict__ bta,
                    _Float16* __restrict__ out, float* __restrict__ raw, int mode) {
    int row = blockIdx.x;
    const float* src;
    if (mode == 0) {
        int bhw = row >> 3, t = row & 7;
        int b = bhw / HW_, hw = bhw % HW_;
        src = x + ((size_t)b * NTOK + 1 + t * HW_ + hw) * C_DIM;
    } else if (mode == 1) {
        int bt = row / 197, j = row % 197;
        int b = bt >> 3, t = bt & 7;
        if (j == 0) src = x + (size_t)b * NTOK * C_DIM;
        else        src = alt + ((size_t)(b * HW_ + (j - 1)) * T_ + t) * C_DIM;
    } else {
        src = x + (size_t)row * C_DIM;
    }
    int tid = threadIdx.x;
    float v0 = src[tid], v1 = src[tid + 256], v2 = src[tid + 512];
    float mean = block_sum256(v0 + v1 + v2) * (1.0f / C_DIM);
    float d0 = v0 - mean, d1 = v1 - mean, d2 = v2 - mean;
    float var = block_sum256(d0 * d0 + d1 * d1 + d2 * d2) * (1.0f / C_DIM);
    float inv = rsqrtf(var + 1e-6f);
    _Float16* op = out + (size_t)row * C_DIM;
    op[tid]       = (_Float16)(d0 * inv * g[tid]       + bta[tid]);
    op[tid + 256] = (_Float16)(d1 * inv * g[tid + 256] + bta[tid + 256]);
    op[tid + 512] = (_Float16)(d2 * inv * g[tid + 512] + bta[tid + 512]);
    if (raw) {
        float* rp = raw + (size_t)row * C_DIM;
        rp[tid] = v0; rp[tid + 256] = v1; rp[tid + 512] = v2;
    }
}

// ------------------------------------------------------------- WMMA GEMM
// C[M,N] = A_f16[M,K] @ W_f16[N,K]^T + bias ; optional residual(f32), GELU,
// f32 and/or f16 outputs.
// Block tile 128(M) x 256(N) x 32(K); 8 waves in 2x4 grid; wave tile 64x64
// = 16 WMMA frags. Double-buffered LDS fed by GLOBAL_LOAD_ASYNC_TO_LDS_B128
// (ASYNCcnt) when available; one workgroup barrier per k-step.
__global__ __launch_bounds__(256)
void gemm_kernel(const _Float16* __restrict__ A, const _Float16* __restrict__ W,
                 const float* __restrict__ bias, const float* __restrict__ res,
                 float* __restrict__ outF, _Float16* __restrict__ outH,
                 int M, int N, int K, int doGelu) {
    __shared__ _Float16 As[2][128][40];   // stride 40 halfs (80B): conflict-free b128
    __shared__ _Float16 Bs[2][256][40];

    int tid = threadIdx.x;
    int blockM = blockIdx.y * 128;
    int blockN = blockIdx.x * 256;
    int lane = tid & 31, wave = tid >> 5;
    int wm = wave >> 2, wn = wave & 3;     // 2 (M) x 4 (N) wave grid
    int l16 = lane & 15, lh = lane >> 4;

    v8f acc[4][4];
    v8f zero = {0.f, 0.f, 0.f, 0.f, 0.f, 0.f, 0.f, 0.f};
    #pragma unroll
    for (int i = 0; i < 4; ++i)
        #pragma unroll
        for (int j = 0; j < 4; ++j) acc[i][j] = zero;

    // loader mapping: A: 128 rows x 32 halfs -> thread pairs; B: 256 rows, 1/thread
    int ldRowA = tid >> 1;
    int ldSegA = (tid & 1) * 16;           // halfs
    int aRow = blockM + ldRowA; if (aRow > M - 1) aRow = M - 1;  // clamp (stores guarded)
    const _Float16* aSrcBase = A + (size_t)aRow * K + ldSegA;
    const _Float16* bSrcBase = W + (size_t)(blockN + tid) * K;   // N multiple of 256

    union Frag { v16h v; float4 f[2]; };
    int ksteps = K >> 5;

#if USE_ASYNC
    // issue tile k into buffer buf (6 x b128 async per thread)
    auto issueTile = [&](int buf, int k0) {
        const _Float16* ag = aSrcBase + k0;
        __builtin_amdgcn_global_load_async_to_lds_b128(
            GP128(ag),     LP128(&As[buf][ldRowA][ldSegA]),     0, 0);
        __builtin_amdgcn_global_load_async_to_lds_b128(
            GP128(ag + 8), LP128(&As[buf][ldRowA][ldSegA + 8]), 0, 0);
        const _Float16* bg = bSrcBase + k0;
        #pragma unroll
        for (int s = 0; s < 4; ++s)
            __builtin_amdgcn_global_load_async_to_lds_b128(
                GP128(bg + s * 8), LP128(&Bs[buf][tid][s * 8]), 0, 0);
    };
    issueTile(0, 0);
    int buf = 0;
    for (int i = 0; i < ksteps; ++i) {
        __builtin_amdgcn_s_wait_asynccnt(0);   // this wave's buf-loads landed in LDS
        __syncthreads();                        // everyone's landed; prev buf free
        if (i + 1 < ksteps) issueTile(buf ^ 1, (i + 1) << 5);
#else
    int buf = 0;
    for (int i = 0; i < ksteps; ++i) {
        int k0 = i << 5;
        __syncthreads();
        {
            const float4* as = reinterpret_cast<const float4*>(aSrcBase + k0);
            float4 a0 = as[0], a1 = as[1];
            *reinterpret_cast<float4*>(&As[0][ldRowA][ldSegA])     = a0;
            *reinterpret_cast<float4*>(&As[0][ldRowA][ldSegA + 8]) = a1;
            const float4* bsrc = reinterpret_cast<const float4*>(bSrcBase + k0);
            float4 b0 = bsrc[0], b1 = bsrc[1], b2 = bsrc[2], b3 = bsrc[3];
            *reinterpret_cast<float4*>(&Bs[0][tid][0])  = b0;
            *reinterpret_cast<float4*>(&Bs[0][tid][8])  = b1;
            *reinterpret_cast<float4*>(&Bs[0][tid][16]) = b2;
            *reinterpret_cast<float4*>(&Bs[0][tid][24]) = b3;
        }
        __syncthreads();
#endif
        // B fragments: lane holds column n=l16, K = 8*lh..+7 then 16+8*lh..+7
        Frag bfr[4];
        #pragma unroll
        for (int fn = 0; fn < 4; ++fn) {
            const _Float16* bp = &Bs[buf][wn * 64 + fn * 16 + l16][0];
            bfr[fn].f[0] = *reinterpret_cast<const float4*>(bp + 8 * lh);
            bfr[fn].f[1] = *reinterpret_cast<const float4*>(bp + 16 + 8 * lh);
        }
        // A fragments: lane holds row m=l16, K = 8*lh..+7 then 16+8*lh..+7
        #pragma unroll
        for (int fm = 0; fm < 4; ++fm) {
            const _Float16* ap = &As[buf][wm * 64 + fm * 16 + l16][0];
            Frag afr;
            afr.f[0] = *reinterpret_cast<const float4*>(ap + 8 * lh);
            afr.f[1] = *reinterpret_cast<const float4*>(ap + 16 + 8 * lh);
            #pragma unroll
            for (int fn = 0; fn < 4; ++fn) {
                acc[fm][fn] = __builtin_amdgcn_wmma_f32_16x16x32_f16(
                    false, afr.v, false, bfr[fn].v, (short)0, acc[fm][fn], false, false);
            }
        }
#if USE_ASYNC
        buf ^= 1;
#endif
    }

    // epilogue: VGPR r of C frag holds m = r + 8*lh, column n = l16
    #pragma unroll
    for (int fm = 0; fm < 4; ++fm) {
        #pragma unroll
        for (int fn = 0; fn < 4; ++fn) {
            int col = blockN + wn * 64 + fn * 16 + l16;
            float bv = bias[col];
            #pragma unroll
            for (int r = 0; r < 8; ++r) {
                int grow = blockM + wm * 64 + fm * 16 + r + 8 * lh;
                if (grow < M) {
                    float v = acc[fm][fn][r] + bv;
                    size_t oi = (size_t)grow * N + col;
                    if (res) v += res[oi];
                    if (doGelu) v = 0.5f * v * (1.0f + erff(v * 0.70710678118f));
                    if (outF) outF[oi] = v;
                    if (outH) outH[oi] = (_Float16)v;
                }
            }
        }
    }
}

// ------------------------------------------------- temporal attention (T=8)
__global__ void attn_t_kernel(const _Float16* __restrict__ qkv, _Float16* __restrict__ o) {
    int bhw = blockIdx.x;
    int tid = threadIdx.x;
    int h = tid >> 3, tq = tid & 7;
    if (h >= NH) return;
    size_t r0 = (size_t)bhw * T_;
    const _Float16* Q = qkv + (r0 + tq) * QKVN + h * HD;
    float q[HD];
    #pragma unroll
    for (int d = 0; d < HD; ++d) q[d] = (float)Q[d];
    float s[T_];
    float m = -1e30f;
    for (int j = 0; j < T_; ++j) {
        const _Float16* Kj = qkv + (r0 + j) * QKVN + C_DIM + h * HD;
        float a = 0.f;
        #pragma unroll
        for (int d = 0; d < HD; ++d) a += q[d] * (float)Kj[d];
        a *= 0.125f;
        s[j] = a; m = fmaxf(m, a);
    }
    float sum = 0.f;
    for (int j = 0; j < T_; ++j) { s[j] = __expf(s[j] - m); sum += s[j]; }
    float inv = 1.0f / sum;
    _Float16* op = o + (r0 + tq) * C_DIM + h * HD;
    for (int d = 0; d < HD; ++d) {
        float a = 0.f;
        for (int j = 0; j < T_; ++j)
            a += s[j] * (float)qkv[(r0 + j) * QKVN + 2 * C_DIM + h * HD + d];
        op[d] = (_Float16)(a * inv);
    }
}

// ------------------------------------------------ spatial attention (L=197)
__global__ __launch_bounds__(256)
void attn_s_kernel(const _Float16* __restrict__ qkv, _Float16* __restrict__ o) {
    __shared__ _Float16 Ks[197 * HD];
    __shared__ _Float16 Vs[197 * HD];
    int bt = blockIdx.x / NH;
    int h  = blockIdx.x % NH;
    int tid = threadIdx.x;
    size_t r0 = (size_t)bt * 197;
    for (int e = tid; e < 197 * HD; e += 256) {
        int j = e / HD, d = e % HD;
        size_t base = (r0 + j) * QKVN + h * HD + d;
        Ks[e] = qkv[base + C_DIM];
        Vs[e] = qkv[base + 2 * C_DIM];
    }
    __syncthreads();
    int wave = tid >> 5, lane = tid & 31;
    for (int tq = wave; tq < 197; tq += 8) {
        const _Float16* Q = qkv + (r0 + tq) * QKVN + h * HD;
        float q[HD];
        #pragma unroll
        for (int d = 0; d < HD; ++d) q[d] = (float)Q[d];
        float sc[7];
        float m = -1e30f;
        int cnt = 0;
        for (int j = lane; j < 197; j += 32) {
            float s = 0.f;
            #pragma unroll
            for (int d = 0; d < HD; ++d) s += q[d] * (float)Ks[j * HD + d];
            s *= 0.125f;
            sc[cnt++] = s;
            m = fmaxf(m, s);
        }
        for (int off = 16; off > 0; off >>= 1) m = fmaxf(m, __shfl_xor(m, off, 32));
        float sum = 0.f;
        float oa[HD];
        #pragma unroll
        for (int d = 0; d < HD; ++d) oa[d] = 0.f;
        cnt = 0;
        for (int j = lane; j < 197; j += 32) {
            float p = __expf(sc[cnt++] - m);
            sum += p;
            #pragma unroll
            for (int d = 0; d < HD; ++d) oa[d] += p * (float)Vs[j * HD + d];
        }
        for (int off = 16; off > 0; off >>= 1) sum += __shfl_xor(sum, off, 32);
        #pragma unroll
        for (int d = 0; d < HD; ++d)
            for (int off = 16; off > 0; off >>= 1) oa[d] += __shfl_xor(oa[d], off, 32);
        float inv = 1.0f / sum;
        _Float16* op = o + (r0 + tq) * C_DIM + h * HD;
        int d0 = lane * 2;
        op[d0]     = (_Float16)(oa[d0] * inv);
        op[d0 + 1] = (_Float16)(oa[d0 + 1] * inv);
    }
}

// -------------------------------------- concat residual + cls-mean combine
__global__ __launch_bounds__(256)
void combine_kernel(const float* __restrict__ x, const float* __restrict__ xt3,
                    const float* __restrict__ res, float* __restrict__ y) {
    int row = blockIdx.x;
    int b = row / NTOK;
    int n = row % NTOK;
    for (int c = threadIdx.x; c < C_DIM; c += 256) {
        float v;
        if (n == 0) {
            float s = 0.f;
            for (int t = 0; t < T_; ++t)
                s += res[((size_t)(b * T_ + t) * 197) * C_DIM + c];
            v = x[(size_t)b * NTOK * C_DIM + c] + s * (1.0f / T_);
        } else {
            int nm = n - 1; int t = nm / HW_; int hw = nm % HW_;
            v = xt3[((size_t)(b * HW_ + hw) * T_ + t) * C_DIM + c]
              + res[((size_t)(b * T_ + t) * 197 + 1 + hw) * C_DIM + c];
        }
        y[(size_t)row * C_DIM + c] = v;
    }
}

// ---------------------------------------------------------------- launcher
extern "C" void kernel_launch(void* const* d_in, const int* in_sizes, int n_in,
                              void* d_out, int out_size, void* d_ws, size_t ws_size,
                              hipStream_t stream) {
    const float* x        = (const float*)d_in[0];
    const float* tn1_g    = (const float*)d_in[1];
    const float* tn1_b    = (const float*)d_in[2];
    const float* t_qkv_w  = (const float*)d_in[3];
    const float* t_qkv_b  = (const float*)d_in[4];
    const float* t_proj_w = (const float*)d_in[5];
    const float* t_proj_b = (const float*)d_in[6];
    const float* t_fc_w   = (const float*)d_in[7];
    const float* t_fc_b   = (const float*)d_in[8];
    const float* n1_g     = (const float*)d_in[9];
    const float* n1_b     = (const float*)d_in[10];
    const float* qkv_w    = (const float*)d_in[11];
    const float* qkv_b    = (const float*)d_in[12];
    const float* proj_w   = (const float*)d_in[13];
    const float* proj_b   = (const float*)d_in[14];
    const float* n2_g     = (const float*)d_in[15];
    const float* n2_b     = (const float*)d_in[16];
    const float* fc1_w    = (const float*)d_in[17];
    const float* fc1_b    = (const float*)d_in[18];
    const float* fc2_w    = (const float*)d_in[19];
    const float* fc2_b    = (const float*)d_in[20];
    float* out = (float*)d_out;

    // ---- workspace carve (256B aligned)
    char* base = (char*)d_ws;
    size_t off = 0;
    auto carve = [&](size_t bytes) -> void* {
        void* p = base + off;
        off = (off + bytes + 255) & ~(size_t)255;
        return p;
    };
    _Float16* WQKV_T  = (_Float16*)carve((size_t)QKVN * C_DIM * 2);
    _Float16* WPROJ_T = (_Float16*)carve((size_t)C_DIM * C_DIM * 2);
    _Float16* WFC_T   = (_Float16*)carve((size_t)C_DIM * C_DIM * 2);
    _Float16* WQKV_S  = (_Float16*)carve((size_t)QKVN * C_DIM * 2);
    _Float16* WPROJ_S = (_Float16*)carve((size_t)C_DIM * C_DIM * 2);
    _Float16* WFC1    = (_Float16*)carve((size_t)FFN * C_DIM * 2);
    _Float16* WFC2    = (_Float16*)carve((size_t)C_DIM * FFN * 2);
    _Float16* LNBUF   = (_Float16*)carve((size_t)M2 * C_DIM * 2);
    _Float16* OBUF    = (_Float16*)carve((size_t)M2 * C_DIM * 2);
    _Float16* QKVB    = (_Float16*)carve((size_t)M2 * QKVN * 2);
    _Float16* X2      = (_Float16*)carve((size_t)M1 * C_DIM * 2);
    _Float16* HB      = (_Float16*)carve((size_t)M3 * FFN * 2);
    float*    XT3     = (float*)   carve((size_t)M1 * C_DIM * 4);
    float*    RES     = (float*)   carve((size_t)M2 * C_DIM * 4);
    float*    YBUF    = (float*)   carve((size_t)M3 * C_DIM * 4); // also raw xt (fp32)

    // ---- weights fp32 -> fp16
    cvt_f32_f16<<<1024, 256, 0, stream>>>(t_qkv_w,  WQKV_T,  QKVN * C_DIM);
    cvt_f32_f16<<<1024, 256, 0, stream>>>(t_proj_w, WPROJ_T, C_DIM * C_DIM);
    cvt_f32_f16<<<1024, 256, 0, stream>>>(t_fc_w,   WFC_T,   C_DIM * C_DIM);
    cvt_f32_f16<<<1024, 256, 0, stream>>>(qkv_w,    WQKV_S,  QKVN * C_DIM);
    cvt_f32_f16<<<1024, 256, 0, stream>>>(proj_w,   WPROJ_S, C_DIM * C_DIM);
    cvt_f32_f16<<<1024, 256, 0, stream>>>(fc1_w,    WFC1,    FFN * C_DIM);
    cvt_f32_f16<<<1024, 256, 0, stream>>>(fc2_w,    WFC2,    C_DIM * FFN);

    // ---- temporal path
    ln_rows_kernel<<<M1, 256, 0, stream>>>(x, nullptr, tn1_g, tn1_b, LNBUF, YBUF, 0);
    gemm_kernel<<<dim3(QKVN / 256, (M1 + 127) / 128), 256, 0, stream>>>(
        LNBUF, WQKV_T, t_qkv_b, nullptr, nullptr, QKVB, M1, QKVN, C_DIM, 0);
    attn_t_kernel<<<B_ * HW_, 96, 0, stream>>>(QKVB, OBUF);
    gemm_kernel<<<dim3(C_DIM / 256, (M1 + 127) / 128), 256, 0, stream>>>(
        OBUF, WPROJ_T, t_proj_b, YBUF, nullptr, X2, M1, C_DIM, C_DIM, 0);
    gemm_kernel<<<dim3(C_DIM / 256, (M1 + 127) / 128), 256, 0, stream>>>(
        X2, WFC_T, t_fc_b, nullptr, XT3, nullptr, M1, C_DIM, C_DIM, 0);

    // ---- spatial path
    ln_rows_kernel<<<M2, 256, 0, stream>>>(x, XT3, n1_g, n1_b, LNBUF, nullptr, 1);
    gemm_kernel<<<dim3(QKVN / 256, (M2 + 127) / 128), 256, 0, stream>>>(
        LNBUF, WQKV_S, qkv_b, nullptr, nullptr, QKVB, M2, QKVN, C_DIM, 0);
    attn_s_kernel<<<B_ * T_ * NH, 256, 0, stream>>>(QKVB, OBUF);
    gemm_kernel<<<dim3(C_DIM / 256, (M2 + 127) / 128), 256, 0, stream>>>(
        OBUF, WPROJ_S, proj_b, nullptr, RES, nullptr, M2, C_DIM, C_DIM, 0);

    // ---- combine + MLP
    combine_kernel<<<M3, 256, 0, stream>>>(x, XT3, RES, YBUF);
    ln_rows_kernel<<<M3, 256, 0, stream>>>(YBUF, nullptr, n2_g, n2_b, LNBUF, nullptr, 2);
    gemm_kernel<<<dim3(FFN / 256, (M3 + 127) / 128), 256, 0, stream>>>(
        LNBUF, WFC1, fc1_b, nullptr, nullptr, HB, M3, FFN, C_DIM, 1);
    gemm_kernel<<<dim3(C_DIM / 256, (M3 + 127) / 128), 256, 0, stream>>>(
        HB, WFC2, fc2_b, YBUF, out, nullptr, M3, C_DIM, FFN, 0);

    (void)in_sizes; (void)n_in; (void)out_size; (void)ws_size;
}
